// VanillaLSTM_82162724372549
// MI455X (gfx1250) — matrix operand order repaired and checked by reading, not compiled, not run on hardware
//
#include <hip/hip_runtime.h>

typedef __attribute__((ext_vector_type(16))) _Float16 v16h;
typedef __attribute__((ext_vector_type(8)))  _Float16 v8h;
typedef __attribute__((ext_vector_type(8)))  float    v8f;

#define BATCH  32768
#define HID    128
#define TOBS   8      // seq-1
#define TPRED  11     // n_predict-1
#define TTOT   (TOBS + TPRED)
#define KPAD   160    // 128 (h) + 16 (x_emb) + 16 (zero pad)
#define NKF    5      // K fragments of 32
#define STRIDE 168    // padded halves per hx row (bank-conflict spread)

#define LOG2E   1.4426950408889634f

// Raw-rate activations: v_exp_f32 / v_rcp_f32, no refinement chains.
__device__ __forceinline__ float sig_fast(float x) {
    return __builtin_amdgcn_rcpf(1.0f + __builtin_amdgcn_exp2f(x * -LOG2E));
}
__device__ __forceinline__ float tanh_fast(float x) {
    return fmaf(-2.0f,
                __builtin_amdgcn_rcpf(__builtin_amdgcn_exp2f(x * (2.0f * LOG2E)) + 1.0f),
                1.0f);
}

// ---------------------------------------------------------------------------
// Prep: pack [W_hh | W_ih | 0] (K=160, N=512) as f16 in WMMA B-fragment order,
// one contiguous 32B chunk per lane per (N-tile, K-frag). Also fuse biases.
// Fragment f = nt*5 + kf; lane l: N = nt*16 + (l&15), K = kf*32 + (l>>4)*16 + i.
// ---------------------------------------------------------------------------
__global__ void lstm_prep(const float* __restrict__ W_ih, const float* __restrict__ b_ih,
                          const float* __restrict__ W_hh, const float* __restrict__ b_hh,
                          _Float16* __restrict__ wsw, float* __restrict__ wsb)
{
    int idx = blockIdx.x * blockDim.x + threadIdx.x;
    if (idx < KPAD * 512) {
        int frag = idx >> 9;          // / (32 lanes * 16 halves)
        int rem  = idx & 511;
        int lane = rem >> 4;
        int i    = rem & 15;
        int nt = frag / NKF;
        int kf = frag % NKF;
        int n = nt * 16 + (lane & 15);
        int k = kf * 32 + (lane >> 4) * 16 + i;
        float v = 0.0f;
        if (k < 128)      v = W_hh[n * 128 + k];         // W_hh^T[k][n]
        else if (k < 144) v = W_ih[n * 16 + (k - 128)];  // W_ih^T[k-128][n]
        wsw[idx] = (_Float16)v;
    }
    if (idx < 512) wsb[idx] = b_ih[idx] + b_hh[idx];
}

// ---------------------------------------------------------------------------
// Persistent LSTM: 256 blocks x 8 waves; each wave owns 16 batch rows for all
// 19 steps. c-state in VGPRs, h-state in wave-private LDS (f16), gates via
// v_wmma_f32_16x16x32_f16 with f32 accumulation. Weight fragments stream from
// L2 each step (pointer laundered so LICM cannot hoist-and-spill them).
// Accumulators start as WMMA with inline-zero C; biases folded into epilogue.
// ---------------------------------------------------------------------------
__global__ __launch_bounds__(256)
void lstm_main(const float* __restrict__ observed,
               const float* __restrict__ W_emb, const float* __restrict__ b_emb,
               const float* __restrict__ W_out, const float* __restrict__ b_out,
               const _Float16* __restrict__ wsw, const float* __restrict__ wsb,
               float* __restrict__ out)
{
    __shared__ __align__(32) _Float16 hx[8 * 16 * STRIDE]; // per-wave [16 rows][K pad 168]
    __shared__ float plbuf[8][16][2];  // last predicted pos
    __shared__ float ppbuf[8][16][2];  // previous predicted pos
    __shared__ float wout_s[2 * HID];
    __shared__ float bout_s[2];

    const int tid  = threadIdx.x;
    const int wave = tid >> 5;
    const int lane = tid & 31;
    const int lrow = lane & 15;   // local batch row / N-column / output row
    const int lhi  = lane >> 4;   // lane half

    const int brow = blockIdx.x * 128 + wave * 16 + lrow;  // global batch row
    _Float16* hxw = &hx[wave * 16 * STRIDE];

    // ---- init: h = 0 (and zero K-pad region), cache readout weights ----
    for (int i = lane; i < 16 * STRIDE; i += 32) hxw[i] = (_Float16)0.0f;
    if (tid < 2 * HID) wout_s[tid] = W_out[tid];
    if (tid < 2)       bout_s[tid] = b_out[tid];
    __syncthreads();

    float cst[8][8];   // c-state: [column-group][C-tile vgpr]
#pragma unroll
    for (int cg = 0; cg < 8; ++cg)
#pragma unroll
        for (int r = 0; r < 8; ++r) cst[cg][r] = 0.0f;

#pragma unroll 1
    for (int t = 0; t < TTOT; ++t) {
        // Launder the weight base pointer: defeats LICM so the 160 B-fragments
        // are loaded from L2 each step instead of being hoisted + spilled.
        const _Float16* wsw_t = wsw;
        asm volatile("" : "+s"(wsw_t));
        const v16h* wb = (const v16h*)wsw_t;

        // ---- per-lane step inputs: embedding delta + position base ----
        float dx0, dx1, base0, base1;
        if (t < TOBS) {
            const float* o0 = observed + ((size_t)t       * BATCH + brow) * 2;
            const float* o1 = observed + ((size_t)(t + 1) * BATCH + brow) * 2;
            float a0 = o0[0], a1 = o0[1], n0 = o1[0], n1 = o1[1];
            dx0 = n0 - a0; dx1 = n1 - a1;
            base0 = n0;    base1 = n1;
        } else {
            float p0 = plbuf[wave][lrow][0], p1 = plbuf[wave][lrow][1];
            float q0 = ppbuf[wave][lrow][0], q1 = ppbuf[wave][lrow][1];
            dx0 = p0 - q0; dx1 = p1 - q1;
            base0 = p0;    base1 = p1;
        }
        float pl_old = plbuf[wave][lrow][lhi];

        // ---- embedding: relu(dx @ W_emb^T + b_emb) -> hx[:, 128..143] ----
#pragma unroll
        for (int e8 = 0; e8 < 8; ++e8) {
            int e = lhi * 8 + e8;
            float v = b_emb[e] + dx0 * W_emb[e * 2 + 0] + dx1 * W_emb[e * 2 + 1];
            v = v > 0.0f ? v : 0.0f;
            hxw[lrow * STRIDE + 128 + e] = (_Float16)v;
        }
        __syncthreads();

        // ---- load A fragments (this wave's 16 rows x K=160 of hx) ----
        v16h afrag[NKF];
        {
            const _Float16* arow = hxw + lrow * STRIDE;
#pragma unroll
            for (int kf = 0; kf < NKF; ++kf) {
                v8h lo  = *(const v8h*)(arow + kf * 32 + lhi * 16);       // K lo 8
                v8h hi8 = *(const v8h*)(arow + kf * 32 + lhi * 16 + 8);  // K hi 8
                afrag[kf] = __builtin_shufflevector(lo, hi8,
                    0, 1, 2, 3, 4, 5, 6, 7, 8, 9, 10, 11, 12, 13, 14, 15);
            }
        }

        // ---- gates + cell update, 8 column groups of 16 hidden units ----
#pragma unroll
        for (int cg = 0; cg < 8; ++cg) {
            v8f acc[4];  // i, f, g, o accumulators (same (lane,vgpr) mapping)
            const v8f zero = {};
            // kf = 0: C operand is the zero vector -> inline SRC2=0, no movs
#pragma unroll
            for (int g = 0; g < 4; ++g) {
                int nt = g * 8 + cg;
                v16h bfrag = wb[(nt * NKF + 0) * 32 + lane];
                acc[g] = __builtin_amdgcn_wmma_f32_16x16x32_f16(
                    false, afrag[0], false, bfrag, (short)0, zero, false, false);
            }
#pragma unroll
            for (int kf = 1; kf < NKF; ++kf) {
#pragma unroll
                for (int g = 0; g < 4; ++g) {
                    int nt = g * 8 + cg;
                    v16h bfrag = wb[(nt * NKF + kf) * 32 + lane];
                    acc[g] = __builtin_amdgcn_wmma_f32_16x16x32_f16(
                        false, afrag[kf], false, bfrag, (short)0, acc[g],
                        false, false);
                }
            }
            // epilogue: add bias scalar, LSTM cell math, h write-back
            float bb0 = wsb[0 * 128 + cg * 16 + lrow];
            float bb1 = wsb[1 * 128 + cg * 16 + lrow];
            float bb2 = wsb[2 * 128 + cg * 16 + lrow];
            float bb3 = wsb[3 * 128 + cg * 16 + lrow];
#pragma unroll
            for (int r = 0; r < 8; ++r) {
                float iv = sig_fast(acc[0][r] + bb0);
                float fv = sig_fast(acc[1][r] + bb1);
                float gv = tanh_fast(acc[2][r] + bb2);
                float ov = sig_fast(acc[3][r] + bb3);
                float c2 = fv * cst[cg][r] + iv * gv;
                float h2 = ov * tanh_fast(c2);
                cst[cg][r] = c2;
                // C layout: row = r + 8*(lane>=16), col = lane&15
                hxw[(r + 8 * lhi) * STRIDE + cg * 16 + lrow] = (_Float16)h2;
            }
        }
        __syncthreads();

        // ---- readout: pos = base + (h @ W_out^T + b_out)[:, lhi] ----
        float s = bout_s[lhi];
        {
            const v8h* hrow8 = (const v8h*)(hxw + lrow * STRIDE);
            const float* wr = wout_s + lhi * HID;
#pragma unroll
            for (int jc = 0; jc < HID / 8; ++jc) {
                v8h hv = hrow8[jc];
#pragma unroll
                for (int u = 0; u < 8; ++u) s += (float)hv[u] * wr[jc * 8 + u];
            }
        }
        float pos = (lhi == 0 ? base0 : base1) + s;
        out[((size_t)t * BATCH + brow) * 2 + lhi] = pos;
        __syncthreads();
        ppbuf[wave][lrow][lhi] = pl_old;   // becomes p_prev next step
        plbuf[wave][lrow][lhi] = pos;      // becomes p_last next step
        __syncthreads();
    }
}

// ---------------------------------------------------------------------------
extern "C" void kernel_launch(void* const* d_in, const int* in_sizes, int n_in,
                              void* d_out, int out_size, void* d_ws, size_t ws_size,
                              hipStream_t stream)
{
    const float* observed = (const float*)d_in[0];
    const float* W_emb    = (const float*)d_in[1];
    const float* b_emb    = (const float*)d_in[2];
    const float* W_ih     = (const float*)d_in[3];
    const float* b_ih     = (const float*)d_in[4];
    const float* W_hh     = (const float*)d_in[5];
    const float* b_hh     = (const float*)d_in[6];
    const float* W_out    = (const float*)d_in[7];
    const float* b_out    = (const float*)d_in[8];
    float* outp = (float*)d_out;

    _Float16* wsw = (_Float16*)d_ws;                       // 160*512 f16 = 160 KB
    float*    wsb = (float*)((char*)d_ws + KPAD * 512 * sizeof(_Float16)); // 512 f32

    lstm_prep<<<(KPAD * 512 + 255) / 256, 256, 0, stream>>>(W_ih, b_ih, W_hh, b_hh, wsw, wsb);
    lstm_main<<<BATCH / 128, 256, 0, stream>>>(observed, W_emb, b_emb, W_out, b_out,
                                               wsw, wsb, outp);
}
